// transformer_block_71390946394579
// MI455X (gfx1250) — matrix operand order repaired and checked by reading, not compiled, not run on hardware
//
#include <hip/hip_runtime.h>
#include <math.h>

// ---------------------------------------------------------------------------
// MI455X (gfx1250) fused residue-transformer block.
// HBM-bound (~1 GB traffic @ 23.3 TB/s dominates compute at WMMA rates):
//  - bf16 intermediates, v_wmma_f32_16x16x32_bf16 for all GEMMs
//  - A-tiles staged to LDS once per workgroup via gfx1250 async loads
//    (global_load_async_to_lds_b128 / s_wait_asynccnt), fragments via ds_load
//  - alpha folded into the attention-out A-fragments
//  - residual + LayerNorm fused into GEMM epilogues through LDS
// ---------------------------------------------------------------------------

#define NN 100000
#define DD 512
#define HH 4
#define SS 256
#define HD (HH * DD)   // 2048
#define FF (2 * DD)    // 1024
#define NT (NN / 16)   // 6250 row tiles

typedef __bf16 bf16_t;
typedef __attribute__((ext_vector_type(8)))  bf16_t v8bf;
typedef __attribute__((ext_vector_type(16))) bf16_t v16bf;
typedef __attribute__((ext_vector_type(8)))  float  v8f;

__device__ __forceinline__ v8f wmma_bf16(v16bf a, v16bf b, v8f c) {
    // D = A(16x32) * B(32x16) + C, f32 accumulate
    return __builtin_amdgcn_wmma_f32_16x16x32_bf16(false, a, false, b,
                                                   (short)0, c, false, false);
}

// A fragment: 16x32 tile of row-major [*, ld] bf16 (global or LDS; the
// address-space is recovered after inlining -> ds_load_b128 for LDS tiles).
// ISA 7.12.2: lane holds row lane%16; K = {koff..+7, koff+16..+23},
// koff = (lane<16 ? 0 : 8).
__device__ __forceinline__ v16bf load_fragA(const bf16_t* base, int ld,
                                            int row0, int k0, int lane) {
    const int r    = lane & 15;
    const int koff = (lane & 16) ? 8 : 0;
    const bf16_t* p = base + (size_t)(row0 + r) * ld + (k0 + koff);
    v8bf lo = *(const v8bf*)p;
    v8bf hi = *(const v8bf*)(p + 16);
    v16bf a;
#pragma unroll
    for (int i = 0; i < 8; ++i) { a[i] = lo[i]; a[i + 8] = hi[i]; }
    return a;
}

// B fragment: 32x16 tile from B-transposed storage Bt[n][k], row-major [*, ld].
// Lane holds column lane%16; K = {koff..+7, koff+16..+23}, koff = (lane<16 ? 0 : 8).
__device__ __forceinline__ v16bf load_fragB(const bf16_t* baseT, int ld,
                                            int col0, int k0, int lane) {
    const int c    = lane & 15;
    const int koff = (lane & 16) ? 8 : 0;
    const bf16_t* p = baseT + (size_t)(col0 + c) * ld + (k0 + koff);
    v8bf lo = *(const v8bf*)p;
    v8bf hi = *(const v8bf*)(p + 16);
    v16bf b;
#pragma unroll
    for (int i = 0; i < 8; ++i) { b[i] = lo[i]; b[i + 8] = hi[i]; }
    return b;
}

// ---------------------------------------------------------------------------
// gfx1250 async global->LDS staging (ASYNCcnt path, ISA ch.10 / 08_async).
// Tile slabs are contiguous (full-width rows), staged in 16B chunks.
// LDS byte offset = low 32 bits of the flat LDS address (ISA 10.2 aperture).
// ---------------------------------------------------------------------------
__device__ __forceinline__ void async_stage(const bf16_t* gsrc, bf16_t* lds_dst,
                                            int nbytes, int tid, int nthreads) {
    const unsigned lbase = (unsigned)(uintptr_t)lds_dst;
    const unsigned long long gbase = (unsigned long long)(uintptr_t)gsrc;
    const int nch = nbytes >> 4;
    for (int c = tid; c < nch; c += nthreads) {
        const unsigned la = lbase + ((unsigned)c << 4);
        const unsigned long long ga = gbase + ((unsigned long long)c << 4);
        asm volatile("global_load_async_to_lds_b128 %0, %1, off"
                     :: "v"(la), "v"(ga) : "memory");
    }
}
__device__ __forceinline__ void async_wait_all() {
    asm volatile("s_wait_asynccnt 0x0" ::: "memory");
}

// ---------------------------------------------------------------------------
// Casting / transposition helpers
// ---------------------------------------------------------------------------
__global__ void cast_f32_bf16_kernel(const float* __restrict__ src,
                                     bf16_t* __restrict__ dst, long n) {
    long i = (long)blockIdx.x * blockDim.x + threadIdx.x;
    const long stride = (long)gridDim.x * blockDim.x;
    for (; i < n; i += stride) dst[i] = (bf16_t)src[i];
}

// Wt[m][f][d] = W[m][d][f], cast to bf16. W: [nmat][rows][cols].
__global__ void transpose_cast_kernel(const float* __restrict__ W,
                                      bf16_t* __restrict__ Wt,
                                      int nmat, int rows, int cols) {
    const long n = (long)nmat * rows * cols;
    long i = (long)blockIdx.x * blockDim.x + threadIdx.x;
    const long stride = (long)gridDim.x * blockDim.x;
    for (; i < n; i += stride) {
        const long mc  = (long)rows * cols;
        const int  m   = (int)(i / mc);
        const long rem = i % mc;
        const int  f   = (int)(rem / rows);
        const int  d   = (int)(rem % rows);
        Wt[i] = (bf16_t)W[(long)m * mc + (long)d * cols + f];
    }
}

// ---------------------------------------------------------------------------
// Per-head q.k score + V (head-major); one wave per (16-row tile, head).
// Xr/Xi rows staged to LDS once (32KB) instead of 32x re-reads per tile.
// ---------------------------------------------------------------------------
__global__ __launch_bounds__(128) void qkv_att_v_kernel(
    const bf16_t* __restrict__ Xr, const bf16_t* __restrict__ Xi,
    const bf16_t* __restrict__ WqT, const bf16_t* __restrict__ WkT,
    const bf16_t* __restrict__ WvT,
    float* __restrict__ att, bf16_t* __restrict__ multi_v) {
    __shared__ bf16_t lxr[16 * DD];
    __shared__ bf16_t lxi[16 * DD];

    const int tid  = threadIdx.x;
    const int lane = tid & 31;
    const int h    = tid >> 5;   // wave id == head
    const int row0 = blockIdx.x * 16;

    async_stage(Xr + (size_t)row0 * DD, lxr, 16 * DD * (int)sizeof(bf16_t), tid, 128);
    async_stage(Xi + (size_t)row0 * DD, lxi, 16 * DD * (int)sizeof(bf16_t), tid, 128);
    async_wait_all();
    __syncthreads();

    const bf16_t* wq = WqT + (size_t)h * DD * DD;
    const bf16_t* wk = WkT + (size_t)h * DD * DD;
    const bf16_t* wv = WvT + (size_t)h * DD * DD;

    float att_acc[8];
#pragma unroll
    for (int i = 0; i < 8; ++i) att_acc[i] = 0.0f;

    const int col   = (lane & 15);
    const int rbase = (lane & 16) ? 8 : 0;

    for (int ct = 0; ct < DD / 16; ++ct) {           // 32 column tiles
        v8f qa = {}; v8f ka = {}; v8f va = {};
        for (int k = 0; k < DD; k += 32) {           // 16 WMMA steps
            v16bf ar = load_fragA(lxr, DD, 0, k, lane);   // ds_load path
            v16bf ai = load_fragA(lxi, DD, 0, k, lane);
            v16bf bq = load_fragB(wq, DD, ct * 16, k, lane);
            v16bf bk = load_fragB(wk, DD, ct * 16, k, lane);
            v16bf bv = load_fragB(wv, DD, ct * 16, k, lane);
            qa = wmma_bf16(ar, bq, qa);
            ka = wmma_bf16(ai, bk, ka);
            va = wmma_bf16(ar, bv, va);
        }
        const int c = ct * 16 + col;
#pragma unroll
        for (int i = 0; i < 8; ++i) {
            const int r = row0 + rbase + i;
            multi_v[(size_t)r * HD + h * DD + c] = (bf16_t)va[i];
            att_acc[i] += qa[i] * ka[i];             // q.k partial (this col)
        }
    }
    // cross-lane reduce over the 16 columns inside each lane-half
#pragma unroll
    for (int i = 0; i < 8; ++i) {
        float v = att_acc[i];
        for (int m = 1; m < 16; m <<= 1) v += __shfl_xor(v, m, 32);
        att_acc[i] = v;
    }
    if ((lane & 15) == 0) {
        const float inv = 0.027950849718747371f;     // 1/sqrt(1280)
#pragma unroll
        for (int i = 0; i < 8; ++i)
            att[(size_t)(row0 + rbase + i) * HH + h] = att_acc[i] * inv;
    }
}

// ---------------------------------------------------------------------------
// Segment bounds (batch is sorted): seg_start[s] = lower_bound(batch, s)
// ---------------------------------------------------------------------------
__global__ void seg_bounds_kernel(const int* __restrict__ batch,
                                  int* __restrict__ seg_start) {
    const int s = blockIdx.x * blockDim.x + threadIdx.x;
    if (s > SS) return;
    int lo = 0, hi = NN;
    while (lo < hi) {
        const int mid = (lo + hi) >> 1;
        if (batch[mid] < s) lo = mid + 1; else hi = mid;
    }
    seg_start[s] = lo;
}

__global__ __launch_bounds__(256) void seg_softmax_kernel(
    const float* __restrict__ att, const int* __restrict__ seg_start,
    float* __restrict__ segmax, float* __restrict__ segsum) {
    const int s  = blockIdx.x;
    const int lo = seg_start[s], hi = seg_start[s + 1];
    __shared__ float red[256];
    for (int h = 0; h < HH; ++h) {
        float m = -3.0e38f;
        for (int i = lo + threadIdx.x; i < hi; i += 256)
            m = fmaxf(m, att[(size_t)i * HH + h]);
        red[threadIdx.x] = m; __syncthreads();
        for (int st = 128; st > 0; st >>= 1) {
            if (threadIdx.x < st)
                red[threadIdx.x] = fmaxf(red[threadIdx.x], red[threadIdx.x + st]);
            __syncthreads();
        }
        const float mm = red[0]; __syncthreads();
        float ssum = 0.0f;
        for (int i = lo + threadIdx.x; i < hi; i += 256)
            ssum += __expf(att[(size_t)i * HH + h] - mm);
        red[threadIdx.x] = ssum; __syncthreads();
        for (int st = 128; st > 0; st >>= 1) {
            if (threadIdx.x < st) red[threadIdx.x] += red[threadIdx.x + st];
            __syncthreads();
        }
        if (threadIdx.x == 0) {
            segmax[s * HH + h] = mm;
            segsum[s * HH + h] = red[0];
        }
        __syncthreads();
    }
}

__global__ void alpha_kernel(const float* __restrict__ att,
                             const int* __restrict__ batch,
                             const float* __restrict__ segmax,
                             const float* __restrict__ segsum,
                             float* __restrict__ alpha) {
    const long idx = (long)blockIdx.x * blockDim.x + threadIdx.x;
    if (idx >= (long)NN * HH) return;
    const int n = (int)(idx / HH), h = (int)(idx % HH);
    const int s = batch[n];
    alpha[idx] = __expf(att[idx] - segmax[s * HH + h]) / segsum[s * HH + h];
}

// ---------------------------------------------------------------------------
// Attention-out GEMM (alpha in A-fragments) + residual + LN -> x(bf16).
// multi_v tile (64KB) async-staged to LDS; LDS reused for the LN tile.
// ---------------------------------------------------------------------------
__global__ __launch_bounds__(256) void attnout_ln_kernel(
    const bf16_t* __restrict__ multi_v, const float* __restrict__ alpha,
    const bf16_t* __restrict__ WcB, const float* __restrict__ residue,
    const float* __restrict__ ln_g, const float* __restrict__ ln_b,
    bf16_t* __restrict__ x_out) {
    __shared__ union SU {
        bf16_t stage[16 * HD];                         // 64KB A-tile stage
        struct {
            float xs[16][DD + 4];
            float rsum[16][16], rsq[16][16];
            float mu[16], rstd[16];
        } e;
    } u;

    const int tid   = threadIdx.x;
    const int lane  = tid & 31;
    const int wid   = tid >> 5;
    const int row0  = blockIdx.x * 16;
    const int fr    = lane & 15;
    const int koff  = (lane & 16) ? 8 : 0;
    const int rbase = (lane & 16) ? 8 : 0;

    async_stage(multi_v + (size_t)row0 * HD, u.stage,
                16 * HD * (int)sizeof(bf16_t), tid, 256);
    float as[HH];
#pragma unroll
    for (int h = 0; h < HH; ++h)
        as[h] = alpha[(size_t)(row0 + fr) * HH + h];
    async_wait_all();
    __syncthreads();

    v8f acc[4] = {};
    for (int subt = 0; subt < 4; ++subt) {
        const int ct = wid * 4 + subt;
        for (int k = 0; k < HD; k += 32) {
            // A = alpha[n, h] * v[n, h*D + f]; h = k/512 is fragment-constant
            const bf16_t* p = u.stage + (size_t)fr * HD + (k + koff);
            v8bf lo = *(const v8bf*)p;
            v8bf hi = *(const v8bf*)(p + 16);
            const float s = as[k >> 9];
            v16bf a;
#pragma unroll
            for (int i = 0; i < 8; ++i) {
                a[i]     = (bf16_t)((float)lo[i] * s);
                a[i + 8] = (bf16_t)((float)hi[i] * s);
            }
            v16bf b = load_fragB(WcB, HD, ct * 16, k, lane);
            acc[subt] = wmma_bf16(a, b, acc[subt]);
        }
    }
    __syncthreads();                  // all stage reads done; reuse LDS

    for (int subt = 0; subt < 4; ++subt) {
        const int c = (wid * 4 + subt) * 16 + fr;
#pragma unroll
        for (int i = 0; i < 8; ++i)
            u.e.xs[rbase + i][c] = acc[subt][i] +
                residue[(size_t)(row0 + rbase + i) * DD + c];
    }
    __syncthreads();

    // LayerNorm: 16 threads per row, 32 cols each
    const int r   = tid >> 4;
    const int sub = tid & 15;
    float sm = 0.0f, sq = 0.0f;
    for (int c = sub * 32; c < sub * 32 + 32; ++c) {
        const float v = u.e.xs[r][c]; sm += v; sq += v * v;
    }
    u.e.rsum[r][sub] = sm; u.e.rsq[r][sub] = sq;
    __syncthreads();
    if (sub == 0) {
        float s1 = 0.0f, s2 = 0.0f;
        for (int j = 0; j < 16; ++j) { s1 += u.e.rsum[r][j]; s2 += u.e.rsq[r][j]; }
        const float m   = s1 / (float)DD;
        const float var = s2 / (float)DD - m * m;
        u.e.mu[r] = m; u.e.rstd[r] = rsqrtf(var + 1e-5f);
    }
    __syncthreads();
    for (int c = sub * 32; c < sub * 32 + 32; ++c) {
        const float v = (u.e.xs[r][c] - u.e.mu[r]) * u.e.rstd[r] * ln_g[c] + ln_b[c];
        x_out[(size_t)(row0 + r) * DD + c] = (bf16_t)v;
    }
}

// ---------------------------------------------------------------------------
// FFN layer 1: h1 = relu(x @ W1.T + b1), bf16 out. x tile staged (16KB).
// ---------------------------------------------------------------------------
__global__ __launch_bounds__(256) void ffn1_kernel(
    const bf16_t* __restrict__ x, const bf16_t* __restrict__ W1B,
    const float* __restrict__ b1, bf16_t* __restrict__ h1) {
    __shared__ bf16_t lx[16 * DD];

    const int tid  = threadIdx.x;
    const int lane = tid & 31;
    const int wid  = tid >> 5;
    const int row0 = blockIdx.x * 16;
    const int ct   = blockIdx.y * 8 + wid;   // 0..63 column tiles of 1024

    async_stage(x + (size_t)row0 * DD, lx, 16 * DD * (int)sizeof(bf16_t), tid, 256);
    async_wait_all();
    __syncthreads();

    v8f acc = {};
    for (int k = 0; k < DD; k += 32) {
        v16bf a = load_fragA(lx, DD, 0, k, lane);
        v16bf b = load_fragB(W1B, DD, ct * 16, k, lane);
        acc = wmma_bf16(a, b, acc);
    }
    const int c     = ct * 16 + (lane & 15);
    const int rbase = (lane & 16) ? 8 : 0;
    const float bias = b1[c];
#pragma unroll
    for (int i = 0; i < 8; ++i)
        h1[(size_t)(row0 + rbase + i) * FF + c] =
            (bf16_t)fmaxf(acc[i] + bias, 0.0f);
}

// ---------------------------------------------------------------------------
// FFN layer 2 + residual + LN -> fp32 output. h1 tile staged (32KB).
// ---------------------------------------------------------------------------
__global__ __launch_bounds__(256) void ffn2_ln_kernel(
    const bf16_t* __restrict__ h1, const bf16_t* __restrict__ W2B,
    const float* __restrict__ b2, const bf16_t* __restrict__ x_in,
    const float* __restrict__ ln_g, const float* __restrict__ ln_b,
    float* __restrict__ out) {
    __shared__ union SU {
        bf16_t stage[16 * FF];                         // 32KB A-tile stage
        struct {
            float xs[16][DD + 4];
            float rsum[16][16], rsq[16][16];
            float mu[16], rstd[16];
        } e;
    } u;

    const int tid   = threadIdx.x;
    const int lane  = tid & 31;
    const int wid   = tid >> 5;
    const int row0  = blockIdx.x * 16;
    const int fr    = lane & 15;
    const int rbase = (lane & 16) ? 8 : 0;

    async_stage(h1 + (size_t)row0 * FF, u.stage,
                16 * FF * (int)sizeof(bf16_t), tid, 256);
    async_wait_all();
    __syncthreads();

    v8f acc[4] = {};
    for (int subt = 0; subt < 4; ++subt) {
        const int ct = wid * 4 + subt;
        for (int k = 0; k < FF; k += 32) {
            v16bf a = load_fragA(u.stage, FF, 0, k, lane);
            v16bf b = load_fragB(W2B, FF, ct * 16, k, lane);
            acc[subt] = wmma_bf16(a, b, acc[subt]);
        }
    }
    __syncthreads();                  // all stage reads done; reuse LDS

    for (int subt = 0; subt < 4; ++subt) {
        const int c = (wid * 4 + subt) * 16 + fr;
        const float bias = b2[c];
#pragma unroll
        for (int i = 0; i < 8; ++i)
            u.e.xs[rbase + i][c] = acc[subt][i] + bias +
                (float)x_in[(size_t)(row0 + rbase + i) * DD + c];
    }
    __syncthreads();

    const int r   = tid >> 4;
    const int sub = tid & 15;
    float sm = 0.0f, sq = 0.0f;
    for (int c = sub * 32; c < sub * 32 + 32; ++c) {
        const float v = u.e.xs[r][c]; sm += v; sq += v * v;
    }
    u.e.rsum[r][sub] = sm; u.e.rsq[r][sub] = sq;
    __syncthreads();
    if (sub == 0) {
        float s1 = 0.0f, s2 = 0.0f;
        for (int j = 0; j < 16; ++j) { s1 += u.e.rsum[r][j]; s2 += u.e.rsq[r][j]; }
        const float m   = s1 / (float)DD;
        const float var = s2 / (float)DD - m * m;
        u.e.mu[r] = m; u.e.rstd[r] = rsqrtf(var + 1e-5f);
    }
    __syncthreads();
    for (int c = sub * 32; c < sub * 32 + 32; ++c) {
        const float v = (u.e.xs[r][c] - u.e.mu[r]) * u.e.rstd[r] * ln_g[c] + ln_b[c];
        out[(size_t)(row0 + r) * DD + c] = v;
    }
}

// ---------------------------------------------------------------------------
// Host launcher
// ---------------------------------------------------------------------------
extern "C" void kernel_launch(void* const* d_in, const int* in_sizes, int n_in,
                              void* d_out, int out_size, void* d_ws,
                              size_t ws_size, hipStream_t stream) {
    const float* residue = (const float*)d_in[0];
    const float* inter   = (const float*)d_in[1];
    const int*   batch   = (const int*)d_in[2];
    const float* Wq      = (const float*)d_in[3];
    const float* Wk      = (const float*)d_in[4];
    const float* Wv      = (const float*)d_in[5];
    const float* Wc      = (const float*)d_in[6];
    const float* W1      = (const float*)d_in[7];
    const float* b1      = (const float*)d_in[8];
    const float* W2      = (const float*)d_in[9];
    const float* b2      = (const float*)d_in[10];
    const float* ln_g    = (const float*)d_in[11];
    const float* ln_b    = (const float*)d_in[12];
    float* out = (float*)d_out;

    char* ws = (char*)d_ws;
    size_t off = 0;
    auto alloc = [&](size_t bytes) -> void* {
        void* p = ws + off;
        off += (bytes + 255) & ~(size_t)255;
        return p;
    };

    bf16_t* Xr      = (bf16_t*)alloc((size_t)NN * DD * sizeof(bf16_t));
    bf16_t* Xi      = (bf16_t*)alloc((size_t)NN * DD * sizeof(bf16_t));
    bf16_t* WqT     = (bf16_t*)alloc((size_t)HH * DD * DD * sizeof(bf16_t));
    bf16_t* WkT     = (bf16_t*)alloc((size_t)HH * DD * DD * sizeof(bf16_t));
    bf16_t* WvT     = (bf16_t*)alloc((size_t)HH * DD * DD * sizeof(bf16_t));
    bf16_t* WcB     = (bf16_t*)alloc((size_t)DD * HD * sizeof(bf16_t));
    bf16_t* W1B     = (bf16_t*)alloc((size_t)FF * DD * sizeof(bf16_t));
    bf16_t* W2B     = (bf16_t*)alloc((size_t)DD * FF * sizeof(bf16_t));
    float*  att     = (float*)alloc((size_t)NN * HH * sizeof(float));
    float*  alpha   = (float*)alloc((size_t)NN * HH * sizeof(float));
    bf16_t* multi_v = (bf16_t*)alloc((size_t)NN * HD * sizeof(bf16_t));
    int*    segst   = (int*)alloc((size_t)(SS + 1) * sizeof(int));
    float*  segmax  = (float*)alloc((size_t)SS * HH * sizeof(float));
    float*  segsum  = (float*)alloc((size_t)SS * HH * sizeof(float));
    bf16_t* x_b     = (bf16_t*)alloc((size_t)NN * DD * sizeof(bf16_t));
    bf16_t* h1      = (bf16_t*)alloc((size_t)NN * FF * sizeof(bf16_t));

    // 1) cast node features + weights to bf16 (transpose Wq/Wk/Wv)
    cast_f32_bf16_kernel<<<2048, 256, 0, stream>>>(residue, Xr, (long)NN * DD);
    cast_f32_bf16_kernel<<<2048, 256, 0, stream>>>(inter, Xi, (long)NN * DD);
    transpose_cast_kernel<<<1024, 256, 0, stream>>>(Wq, WqT, HH, DD, DD);
    transpose_cast_kernel<<<1024, 256, 0, stream>>>(Wk, WkT, HH, DD, DD);
    transpose_cast_kernel<<<1024, 256, 0, stream>>>(Wv, WvT, HH, DD, DD);
    cast_f32_bf16_kernel<<<1024, 256, 0, stream>>>(Wc, WcB, (long)DD * HD);
    cast_f32_bf16_kernel<<<1024, 256, 0, stream>>>(W1, W1B, (long)FF * DD);
    cast_f32_bf16_kernel<<<1024, 256, 0, stream>>>(W2, W2B, (long)DD * FF);

    // 2) per-head scores + V
    qkv_att_v_kernel<<<NT, 128, 0, stream>>>(Xr, Xi, WqT, WkT, WvT, att, multi_v);

    // 3) segment softmax (batch sorted -> binary-searched ranges)
    seg_bounds_kernel<<<2, 256, 0, stream>>>(batch, segst);
    seg_softmax_kernel<<<SS, 256, 0, stream>>>(att, segst, segmax, segsum);
    alpha_kernel<<<((long)NN * HH + 255) / 256, 256, 0, stream>>>(
        att, batch, segmax, segsum, alpha);

    // 4) attention out + residual + LN
    attnout_ln_kernel<<<NT, 256, 0, stream>>>(multi_v, alpha, WcB, residue,
                                              ln_g, ln_b, x_b);

    // 5) FFN + residual + LN
    ffn1_kernel<<<dim3(NT, FF / 128), 256, 0, stream>>>(x_b, W1B, b1, h1);
    ffn2_ln_kernel<<<NT, 256, 0, stream>>>(h1, W2B, b2, x_b, ln_g, ln_b, out);

    (void)in_sizes; (void)n_in; (void)out_size; (void)ws_size;
}
